// Siglip2TextTransformer_4303557231416
// MI455X (gfx1250) — matrix operand order repaired and checked by reading, not compiled, not run on hardware
//
#include <hip/hip_runtime.h>

// ---------------------------------------------------------------------------
// Siglip2 text transformer forward for MI455X (gfx1250, wave32, WMMA).
// Dense math -> v_wmma_f32_16x16x32_bf16; GEMM tile fill -> async global->LDS
// (ASYNCcnt) via global_load_async_to_lds_b128.
// ---------------------------------------------------------------------------

typedef unsigned short u16;
typedef __attribute__((ext_vector_type(16))) __bf16 bf16x16;
typedef __attribute__((ext_vector_type(8)))  float  f32x8;
typedef __attribute__((ext_vector_type(4)))  int    v4i;

namespace cfg {
constexpr int B = 256, S = 64, D = 768, H = 12, L = 12, F = 3072, P = 768;
constexpr int M = B * S;   // 16384 token rows
}

__device__ __forceinline__ u16 f2bf(float f) {
  unsigned u = __float_as_uint(f);
  u += 0x7FFFu + ((u >> 16) & 1u);       // round-to-nearest-even
  return (u16)(u >> 16);
}

// ---- gfx1250 async global->LDS copy (direct to LDS, tracked by ASYNCcnt) ---
#if __has_builtin(__builtin_amdgcn_global_load_async_to_lds_b128) && \
    __has_builtin(__builtin_amdgcn_s_wait_asynccnt)
#define HAVE_ASYNC_LDS 1
typedef __attribute__((address_space(1))) v4i gv4i;   // global int4
typedef __attribute__((address_space(3))) v4i lv4i;   // LDS int4
__device__ __forceinline__ void async_copy16(const u16* g, u16* l) {
  __builtin_amdgcn_global_load_async_to_lds_b128(
      (gv4i*)(g), (lv4i*)(l), /*offset=*/0, /*cpol=*/0);
}
#else
#define HAVE_ASYNC_LDS 0
#endif

// ---------------------------------------------------------------------------
// Weight transpose + f32 -> bf16 convert: w[K][N] -> wt[N][K] (per blockIdx.z
// matrix). Row-major-in-K output makes WMMA B-fragments contiguous.
// ---------------------------------------------------------------------------
__global__ __launch_bounds__(256) void wtransp_kernel(
    const float* __restrict__ w, u16* __restrict__ wt, int K, int N) {
  __shared__ float tile[32][33];
  w  += (size_t)blockIdx.z * K * N;
  wt += (size_t)blockIdx.z * N * K;
  const int tx = threadIdx.x & 31, ty = threadIdx.x >> 5;
#pragma unroll
  for (int i = 0; i < 4; ++i) {
    int k = blockIdx.y * 32 + ty + i * 8;
    tile[ty + i * 8][tx] = w[(size_t)k * N + blockIdx.x * 32 + tx];
  }
  __syncthreads();
#pragma unroll
  for (int i = 0; i < 4; ++i) {
    int n = blockIdx.x * 32 + ty + i * 8;
    wt[(size_t)n * K + blockIdx.y * 32 + tx] = f2bf(tile[tx][ty + i * 8]);
  }
}

// ---------------------------------------------------------------------------
// Embedding: h[b,s,:] = tok_emb[ids[b,s],:] + pos_emb[s,:]
// ---------------------------------------------------------------------------
__global__ __launch_bounds__(256) void embed_kernel(
    const int* __restrict__ ids, const float* __restrict__ tok,
    const float* __restrict__ pos, float* __restrict__ h) {
  size_t i = (size_t)blockIdx.x * 256 + threadIdx.x;   // < M*D
  int d = (int)(i % cfg::D);
  size_t bs = i / cfg::D;
  int s = (int)(bs % cfg::S);
  h[i] = tok[(size_t)ids[bs] * cfg::D + d] + pos[(size_t)s * cfg::D + d];
}

// ---------------------------------------------------------------------------
// LayerNorm over D=768: one wave32 per row, 24 elements per lane,
// shfl_xor wave reduction. Optionally writes bf16 (GEMM input) and/or f32.
// ---------------------------------------------------------------------------
__global__ __launch_bounds__(256) void layernorm_kernel(
    const float* __restrict__ x, const float* __restrict__ gamma,
    const float* __restrict__ beta, u16* __restrict__ outB,
    float* __restrict__ outF, int Mrows) {
  const int wave = threadIdx.x >> 5, lane = threadIdx.x & 31;
  const int row = blockIdx.x * 8 + wave;
  if (row >= Mrows) return;
  const float* xr = x + (size_t)row * cfg::D;
  float vbuf[24];
  float s = 0.f, sq = 0.f;
#pragma unroll
  for (int i = 0; i < 24; ++i) {
    float v = xr[lane + i * 32];
    vbuf[i] = v; s += v; sq += v * v;
  }
#pragma unroll
  for (int o = 16; o > 0; o >>= 1) {
    s  += __shfl_xor(s, o, 32);
    sq += __shfl_xor(sq, o, 32);
  }
  const float mean = s * (1.f / cfg::D);
  const float var  = sq * (1.f / cfg::D) - mean * mean;
  const float rs   = rsqrtf(var + 1e-6f);
#pragma unroll
  for (int i = 0; i < 24; ++i) {
    int d = lane + i * 32;
    float y = (vbuf[i] - mean) * rs * gamma[d] + beta[d];
    if (outB) outB[(size_t)row * cfg::D + d] = f2bf(y);
    if (outF) outF[(size_t)row * cfg::D + d] = y;
  }
}

// ---------------------------------------------------------------------------
// Attention: one block of 64 threads per (b, h). S=64, HD=64.
// K/V staged in LDS, scores in LDS, q row + output row in registers.
// ~1.4% of total FLOPs -> VALU is the right tool; writes bf16 for O-proj GEMM.
// ---------------------------------------------------------------------------
__global__ __launch_bounds__(64) void attn_kernel(
    const float* __restrict__ q, const float* __restrict__ k,
    const float* __restrict__ v, const int* __restrict__ amask,
    u16* __restrict__ out) {
  __shared__ float Ks[64][64];
  __shared__ float Vs[64][64];
  __shared__ float Sc[64][65];           // +1 pad: conflict-free row access
  const int bh = blockIdx.x;
  const int b = bh / cfg::H, h = bh % cfg::H;
  const int t = threadIdx.x;             // query row / loader row
  const size_t base = (size_t)b * cfg::S * cfg::D + (size_t)h * 64;
#pragma unroll 4
  for (int d = 0; d < 64; ++d) {
    Ks[t][d] = k[base + (size_t)t * cfg::D + d];
    Vs[t][d] = v[base + (size_t)t * cfg::D + d];
  }
  __syncthreads();
  float qr[64];
#pragma unroll
  for (int d = 0; d < 64; ++d)
    qr[d] = q[base + (size_t)t * cfg::D + d] * 0.125f;   // 1/sqrt(64)
  float mx = -3.4e38f;
  for (int kk = 0; kk < 64; ++kk) {
    float acc = 0.f;
#pragma unroll
    for (int d = 0; d < 64; ++d) acc += qr[d] * Ks[kk][d];
    if (amask[b * cfg::S + kk] == 0) acc = -3.4e38f;
    Sc[t][kk] = acc;
    mx = fmaxf(mx, acc);
  }
  float sum = 0.f;
  for (int kk = 0; kk < 64; ++kk) {
    float e = __expf(Sc[t][kk] - mx);
    Sc[t][kk] = e;
    sum += e;
  }
  const float inv = 1.f / sum;
  float o[64];
#pragma unroll
  for (int d = 0; d < 64; ++d) o[d] = 0.f;
  for (int kk = 0; kk < 64; ++kk) {
    float p = Sc[t][kk] * inv;
#pragma unroll
    for (int d = 0; d < 64; ++d) o[d] += p * Vs[kk][d];
  }
#pragma unroll
  for (int d = 0; d < 64; ++d)
    out[base + (size_t)t * cfg::D + d] = f2bf(o[d]);
}

// ---------------------------------------------------------------------------
// WMMA GEMM: Y[M,N] = A[M,K] @ W[K,N] + bias (+ fused epilogue).
// A bf16 row-major, W supplied transposed (Wt[N][K], bf16).
// Block = 256 threads = 8 wave32s; tile 128x128x32; wave tile 32x64
// (2x4 grid of 16x16 accumulators). v_wmma_f32_16x16x32_bf16.
// Tile fill: global_load_async_to_lds_b128 (ASYNCcnt) when available.
// ---------------------------------------------------------------------------
constexpr int EPI_F32 = 0;        // bias, write f32
constexpr int EPI_GELU_BF16 = 1;  // bias + tanh-GELU, write bf16
constexpr int EPI_RES_F32 = 2;    // bias + residual add, write f32

union Frag32B { uint4 u[2]; bf16x16 v; };

__device__ __forceinline__ float gelu_tanh(float x) {
  return 0.5f * x * (1.f + tanhf(0.7978845608f * (x + 0.044715f * x * x * x)));
}

template <int EPI>
__global__ __launch_bounds__(256) void gemm_bf16_kernel(
    const u16* __restrict__ A, const u16* __restrict__ Wt,
    const float* __restrict__ bias, const float* __restrict__ resid,
    float* __restrict__ outF, u16* __restrict__ outB, int N, int K) {
  constexpr int BKP = 40;  // 32 + 8 halfword pad: kills 4-way LDS bank conflict
  __shared__ __align__(16) u16 As[128][BKP];
  __shared__ __align__(16) u16 Bs[128][BKP];
  const int tid = threadIdx.x;
  const int wave = tid >> 5, lane = tid & 31;
  const int wr = wave >> 1, wc = wave & 1;   // 4x2 wave grid
  const int half = lane >> 4, r = lane & 15;
  const int m0 = blockIdx.y * 128, n0 = blockIdx.x * 128;

  // Per-thread tile-fill coordinates: row slab pair 64 rows apart, 16B column.
  const int rowf = tid >> 2;           // 0..63
  const int cf   = (tid & 3) * 8;      // halfword offset 0/8/16/24
  const u16* gA = A  + (size_t)(m0 + rowf) * K + cf;
  const u16* gB = Wt + (size_t)(n0 + rowf) * K + cf;
  const size_t rstep = (size_t)64 * K;
  u16* lA0 = &As[rowf][cf];
  u16* lA1 = &As[rowf + 64][cf];
  u16* lB0 = &Bs[rowf][cf];
  u16* lB1 = &Bs[rowf + 64][cf];

  const f32x8 zero = {0.f, 0.f, 0.f, 0.f, 0.f, 0.f, 0.f, 0.f};
  f32x8 acc[2][4];
#pragma unroll
  for (int i = 0; i < 2; ++i)
#pragma unroll
    for (int j = 0; j < 4; ++j) acc[i][j] = zero;

  for (int kt = 0; kt < K; kt += 32) {
#if HAVE_ASYNC_LDS
    // Direct global->LDS DMA: no VGPR staging, all four 16B lines in flight.
    async_copy16(gA + kt,         lA0);
    async_copy16(gA + rstep + kt, lA1);
    async_copy16(gB + kt,         lB0);
    async_copy16(gB + rstep + kt, lB1);
#else
    // Fallback: issue all global loads before any LDS store (single wait).
    uint4 t0 = *reinterpret_cast<const uint4*>(gA + kt);
    uint4 t1 = *reinterpret_cast<const uint4*>(gA + rstep + kt);
    uint4 t2 = *reinterpret_cast<const uint4*>(gB + kt);
    uint4 t3 = *reinterpret_cast<const uint4*>(gB + rstep + kt);
    *reinterpret_cast<uint4*>(lA0) = t0;
    *reinterpret_cast<uint4*>(lA1) = t1;
    *reinterpret_cast<uint4*>(lB0) = t2;
    *reinterpret_cast<uint4*>(lB1) = t3;
#endif
    if (kt + 32 < K) {  // stream next K-slab into L2 (global_prefetch_b8)
      __builtin_prefetch(gA + kt + 32, 0, 3);
      __builtin_prefetch(gB + kt + 32, 0, 3);
    }
#if HAVE_ASYNC_LDS
    __builtin_amdgcn_s_wait_asynccnt(0);   // this wave's DMA done
#endif
    __syncthreads();                       // whole block's tile visible

    // A-fragment (ISA 16-bit A 16x32 layout): lane<16 holds K 0..7 & 16..23,
    // lane>=16 holds K 8..15 & 24..31, row = lane%16.
    Frag32B a[2], b[4];
#pragma unroll
    for (int i = 0; i < 2; ++i) {
      int row = wr * 32 + i * 16 + r;
      a[i].u[0] = *reinterpret_cast<const uint4*>(&As[row][half * 8]);
      a[i].u[1] = *reinterpret_cast<const uint4*>(&As[row][16 + half * 8]);
    }
    // B-fragment (K x N, 16-bit): lanes 0-15 hold K 0..7 & 16..23, lanes 16-31
    // hold K 8..15 & 24..31, column = lane%16 (same K order as A).
#pragma unroll
    for (int j = 0; j < 4; ++j) {
      int col = wc * 64 + j * 16 + r;
      b[j].u[0] = *reinterpret_cast<const uint4*>(&Bs[col][half * 8]);
      b[j].u[1] = *reinterpret_cast<const uint4*>(&Bs[col][16 + half * 8]);
    }
#pragma unroll
    for (int i = 0; i < 2; ++i)
#pragma unroll
      for (int j = 0; j < 4; ++j)
        acc[i][j] = __builtin_amdgcn_wmma_f32_16x16x32_bf16(
            false, a[i].v, false, b[j].v, (short)0, acc[i][j], false, false);
    __syncthreads();
  }

  // Epilogue. C/D layout: elem g, lane l -> row = g + (l<16?0:8), col = l%16.
#pragma unroll
  for (int i = 0; i < 2; ++i)
#pragma unroll
    for (int j = 0; j < 4; ++j) {
      const int col = n0 + wc * 64 + j * 16 + r;
      const float bval = bias[col];
#pragma unroll
      for (int g = 0; g < 8; ++g) {
        const int row = m0 + wr * 32 + i * 16 + g + half * 8;
        float y = acc[i][j][g] + bval;
        if constexpr (EPI == EPI_GELU_BF16) y = gelu_tanh(y);
        if constexpr (EPI == EPI_RES_F32) y += resid[(size_t)row * N + col];
        if constexpr (EPI == EPI_GELU_BF16)
          outB[(size_t)row * N + col] = f2bf(y);
        else
          outF[(size_t)row * N + col] = y;
      }
    }
}

// ---------------------------------------------------------------------------
// Gather EOS rows (s = S-1) of last_hidden (f32) -> bf16 head-GEMM input.
// ---------------------------------------------------------------------------
__global__ __launch_bounds__(256) void pool_gather_kernel(
    const float* __restrict__ lh, u16* __restrict__ p) {
  int i = blockIdx.x * 256 + threadIdx.x;      // < B*D
  int b = i / cfg::D, d = i % cfg::D;
  p[i] = f2bf(lh[((size_t)b * cfg::S + (cfg::S - 1)) * cfg::D + d]);
}

// ---------------------------------------------------------------------------
// Host orchestration
// ---------------------------------------------------------------------------
extern "C" void kernel_launch(void* const* d_in, const int* in_sizes, int n_in,
                              void* d_out, int out_size, void* d_ws,
                              size_t ws_size, hipStream_t stream) {
  using namespace cfg;
  (void)in_sizes; (void)n_in; (void)out_size; (void)ws_size;

  const int*   ids   = (const int*)d_in[0];
  const int*   amask = (const int*)d_in[1];
  const float* tok   = (const float*)d_in[2];
  const float* pos   = (const float*)d_in[3];
  const float* ln1g  = (const float*)d_in[4];
  const float* ln1b  = (const float*)d_in[5];
  const float* wq    = (const float*)d_in[6];
  const float* bq    = (const float*)d_in[7];
  const float* wk    = (const float*)d_in[8];
  const float* bk    = (const float*)d_in[9];
  const float* wv    = (const float*)d_in[10];
  const float* bv    = (const float*)d_in[11];
  const float* wo    = (const float*)d_in[12];
  const float* bo    = (const float*)d_in[13];
  const float* ln2g  = (const float*)d_in[14];
  const float* ln2b  = (const float*)d_in[15];
  const float* w1    = (const float*)d_in[16];
  const float* b1    = (const float*)d_in[17];
  const float* w2    = (const float*)d_in[18];
  const float* b2    = (const float*)d_in[19];
  const float* flng  = (const float*)d_in[20];
  const float* flnb  = (const float*)d_in[21];
  const float* hw    = (const float*)d_in[22];
  const float* hb    = (const float*)d_in[23];

  // Bump allocator over workspace.
  char* ws = (char*)d_ws;
  auto alloc = [&](size_t bytes) -> void* {
    void* p = ws;
    ws += (bytes + 255) & ~(size_t)255;
    return p;
  };
  u16*   wqt  = (u16*)alloc((size_t)L * D * D * 2);
  u16*   wkt  = (u16*)alloc((size_t)L * D * D * 2);
  u16*   wvt  = (u16*)alloc((size_t)L * D * D * 2);
  u16*   wot  = (u16*)alloc((size_t)L * D * D * 2);
  u16*   w1t  = (u16*)alloc((size_t)L * D * F * 2);
  u16*   w2t  = (u16*)alloc((size_t)L * F * D * 2);
  u16*   hwt  = (u16*)alloc((size_t)D * P * 2);
  float* hbuf = (float*)alloc((size_t)M * D * 4);   // residual stream (f32)
  u16*   xb   = (u16*)alloc((size_t)M * D * 2);     // LN output (bf16)
  float* qbuf = (float*)alloc((size_t)M * D * 4);
  float* kbuf = (float*)alloc((size_t)M * D * 4);
  float* vbuf = (float*)alloc((size_t)M * D * 4);
  u16*   abuf = (u16*)alloc((size_t)M * D * 2);     // attention out (bf16)
  u16*   mbuf = (u16*)alloc((size_t)M * F * 2);     // MLP intermediate (bf16)
  u16*   pbuf = (u16*)alloc((size_t)B * D * 2);     // pooled rows (bf16)

  // ---- one-time weight transpose + bf16 convert ----
  wtransp_kernel<<<dim3(D / 32, D / 32, L), 256, 0, stream>>>(wq, wqt, D, D);
  wtransp_kernel<<<dim3(D / 32, D / 32, L), 256, 0, stream>>>(wk, wkt, D, D);
  wtransp_kernel<<<dim3(D / 32, D / 32, L), 256, 0, stream>>>(wv, wvt, D, D);
  wtransp_kernel<<<dim3(D / 32, D / 32, L), 256, 0, stream>>>(wo, wot, D, D);
  wtransp_kernel<<<dim3(F / 32, D / 32, L), 256, 0, stream>>>(w1, w1t, D, F);
  wtransp_kernel<<<dim3(D / 32, F / 32, L), 256, 0, stream>>>(w2, w2t, F, D);
  wtransp_kernel<<<dim3(P / 32, D / 32, 1), 256, 0, stream>>>(hw, hwt, D, P);

  // ---- embeddings ----
  embed_kernel<<<(size_t)M * D / 256, 256, 0, stream>>>(ids, tok, pos, hbuf);

  const dim3 gD(D / 128, M / 128);   // GEMMs with N = D
  const dim3 gF(F / 128, M / 128);   // GEMM with N = F

  for (int l = 0; l < L; ++l) {
    const size_t oDD = (size_t)l * D * D;
    const size_t oDF = (size_t)l * D * F;
    // LN1 -> bf16
    layernorm_kernel<<<M / 8, 256, 0, stream>>>(hbuf, ln1g + l * D,
                                                ln1b + l * D, xb, nullptr, M);
    // QKV projections
    gemm_bf16_kernel<EPI_F32><<<gD, 256, 0, stream>>>(
        xb, wqt + oDD, bq + l * D, nullptr, qbuf, nullptr, D, D);
    gemm_bf16_kernel<EPI_F32><<<gD, 256, 0, stream>>>(
        xb, wkt + oDD, bk + l * D, nullptr, kbuf, nullptr, D, D);
    gemm_bf16_kernel<EPI_F32><<<gD, 256, 0, stream>>>(
        xb, wvt + oDD, bv + l * D, nullptr, vbuf, nullptr, D, D);
    // Attention
    attn_kernel<<<B * H, 64, 0, stream>>>(qbuf, kbuf, vbuf, amask, abuf);
    // O-projection + residual
    gemm_bf16_kernel<EPI_RES_F32><<<gD, 256, 0, stream>>>(
        abuf, wot + oDD, bo + l * D, hbuf, hbuf, nullptr, D, D);
    // LN2 -> bf16
    layernorm_kernel<<<M / 8, 256, 0, stream>>>(hbuf, ln2g + l * D,
                                                ln2b + l * D, xb, nullptr, M);
    // MLP up + GELU -> bf16
    gemm_bf16_kernel<EPI_GELU_BF16><<<gF, 256, 0, stream>>>(
        xb, w1t + oDF, b1 + l * F, nullptr, nullptr, mbuf, F, D);
    // MLP down + residual
    gemm_bf16_kernel<EPI_RES_F32><<<gD, 256, 0, stream>>>(
        mbuf, w2t + oDF, b2 + l * D, hbuf, hbuf, nullptr, D, F);
  }

  // Final LN -> last_hidden (f32) directly into d_out
  float* last_hidden = (float*)d_out;
  layernorm_kernel<<<M / 8, 256, 0, stream>>>(hbuf, flng, flnb, nullptr,
                                              last_hidden, M);
  // EOS pooling + head projection -> d_out tail
  pool_gather_kernel<<<B * D / 256, 256, 0, stream>>>(last_hidden, pbuf);
  float* pooled = last_hidden + (size_t)M * D;
  gemm_bf16_kernel<EPI_F32><<<dim3(P / 128, B / 128), 256, 0, stream>>>(
      pbuf, hwt, hb, nullptr, pooled, nullptr, P, D);
}
